// GRU_73177652789692
// MI455X (gfx1250) — matrix operand order repaired and checked by reading, not compiled, not run on hardware
//
#include <hip/hip_runtime.h>
#include <stdint.h>

#define B_ 64
#define T_ 2048
#define E_ 256
#define H_ 256
#define ROWP 264   // padded LDS row stride in bf16 elements (multiple of 8 for 16B align)

typedef __attribute__((ext_vector_type(16))) __bf16 bf16x16;
typedef __attribute__((ext_vector_type(8)))  float  f32x8;

union BfVec { bf16x16 v; uint4 q[2]; };

__device__ __forceinline__ unsigned short f2bf(float f) {
  union { float f; unsigned u; } c; c.f = f;
  unsigned u = c.u;
  u = (u + 0x7fffu + ((u >> 16) & 1u)) >> 16;   // round-to-nearest-even
  return (unsigned short)u;
}

// ---------------- Phase 0a: x f32 -> bf16 ----------------
__global__ void k_conv_x(const float* __restrict__ x, unsigned short* __restrict__ xb) {
  size_t i = ((size_t)blockIdx.x * blockDim.x + threadIdx.x) * 8;
  float4 a = *(const float4*)(x + i);
  float4 b = *(const float4*)(x + i + 4);
  union { unsigned short s[8]; uint4 q; } o;
  o.s[0]=f2bf(a.x); o.s[1]=f2bf(a.y); o.s[2]=f2bf(a.z); o.s[3]=f2bf(a.w);
  o.s[4]=f2bf(b.x); o.s[5]=f2bf(b.y); o.s[6]=f2bf(b.z); o.s[7]=f2bf(b.w);
  *(uint4*)(xb + i) = o.q;
}

// ---------------- Phase 0b: split + transpose + convert weights ----------------
// Wg is [H][E+H] f32.  Outputs: wxT[g][n][k] = Wg[n][k] (k<E), whT[g][n][k] = Wg[n][E+k]
__global__ void k_conv_w(const float* __restrict__ Wu, const float* __restrict__ Wr,
                         const float* __restrict__ Wn,
                         unsigned short* __restrict__ wxT, unsigned short* __restrict__ whT) {
  int idx = blockIdx.x * blockDim.x + threadIdx.x;
  const int per = H_ * (E_ + H_);
  int g = idx / per, r = idx % per;
  int n = r / (E_ + H_), c = r % (E_ + H_);
  const float* W = (g == 0) ? Wu : ((g == 1) ? Wr : Wn);
  unsigned short v = f2bf(W[(size_t)n * (E_ + H_) + c]);
  if (c < E_) wxT[(size_t)g * E_ * H_ + (size_t)n * E_ + c] = v;
  else        whT[(size_t)g * H_ * H_ + (size_t)n * H_ + (c - E_)] = v;
}

// ---------------- Phase 1: xg[g][t][b][n] = x @ WgxT + bg ----------------
// Each wave: one 16-row M tile x full N=256, K=256. grid=(M/16/4, 3), block=128.
// 8 physically distinct B buffers, each reloaded right after use and consumed
// 8 WMMAs later -> forces deep load pipelining (partial loadcnt waits).
__global__ void __launch_bounds__(128)
k_proj(const unsigned short* __restrict__ xb, const unsigned short* __restrict__ wxT,
       const float* __restrict__ bu, const float* __restrict__ br,
       const float* __restrict__ bn, float* __restrict__ xg) {
  const int lane = threadIdx.x & 31;
  const int wave = threadIdx.x >> 5;
  const int gate = blockIdx.y;
  const int m0   = (blockIdx.x * 4 + wave) * 16;
  const float* bias = (gate == 0) ? bu : ((gate == 1) ? br : bn);
  const unsigned short* wT = wxT + (size_t)gate * E_ * H_;
  float* out = xg + (size_t)gate * T_ * B_ * H_;

  const int lrow  = lane & 15;
  const int khalf = lane >> 4;

  f32x8 acc[16];
  #pragma unroll
  for (int nt = 0; nt < 16; ++nt) acc[nt] = {};

  const unsigned short* abase = xb + (size_t)(m0 + lrow) * E_ + 8 * khalf;
  auto loadA = [&](int ks) -> bf16x16 {
    BfVec v; const unsigned short* ap = abase + 32 * ks;
    v.q[0] = *(const uint4*)(ap); v.q[1] = *(const uint4*)(ap + 16);
    return v.v;
  };
  auto loadB = [&](int ks, int nt) -> bf16x16 {
    int krow = 8 * khalf + 32 * ks;
    BfVec v; const unsigned short* bp = wT + (size_t)(nt * 16 + lrow) * E_ + krow;
    v.q[0] = *(const uint4*)(bp); v.q[1] = *(const uint4*)(bp + 16);
    return v.v;
  };

  bf16x16 Av[2];
  bf16x16 Bv[8];
  Av[0] = loadA(0);
  #pragma unroll
  for (int j = 0; j < 8; ++j) Bv[j] = loadB(0, j);

  #pragma unroll
  for (int i = 0; i < 128; ++i) {
    const int ks = i >> 4, nt = i & 15;
    acc[nt] = __builtin_amdgcn_wmma_f32_16x16x32_bf16(
        false, Av[ks & 1], false, Bv[i & 7], (short)0, acc[nt], false, false);
    const int nx = i + 8;                       // refill slot just consumed
    if (nx < 128) Bv[i & 7] = loadB(nx >> 4, nx & 15);
    if ((i & 15) == 8 && ks < 7) Av[(ks + 1) & 1] = loadA(ks + 1);
  }

  // epilogue: add bias, scatter to time-major staging xg[t][b][n], M = b*T + t
  #pragma unroll
  for (int nt = 0; nt < 16; ++nt) {
    int n = nt * 16 + lrow;
    float bv = bias[n];
    #pragma unroll
    for (int r = 0; r < 8; ++r) {
      int M = m0 + r + 8 * khalf;
      int b = M >> 11, t = M & (T_ - 1);
      out[((size_t)t * B_ + b) * H_ + n] = acc[nt][r] + bv;
    }
  }
}

// ---------------- Phase 2: recurrent scan ----------------
// 4 workgroups (batch groups of 16) x 512 threads (16 waves).
// Wave w owns columns n0=16w; computes all 3 gate tiles so combine is wave-local.
// Gates u,r weights live in VGPRs; gate n streamed from L2 via 4-slot rotation,
// loaded 3 K-steps (9 WMMAs) ahead of use.
__global__ void __launch_bounds__(512)
k_rnn(const unsigned short* __restrict__ whT, const float* __restrict__ xg,
      const float* __restrict__ h0, float* __restrict__ out) {
  __shared__ unsigned short hl[2][16 * ROWP];   // double-buffered h tile (bf16)

  const int lane  = threadIdx.x & 31;
  const int wave  = threadIdx.x >> 5;
  const int b0    = blockIdx.x * 16;
  const int n0    = wave * 16;
  const int lrow  = lane & 15;
  const int khalf = lane >> 4;

  const unsigned short* wnT = whT + (size_t)2 * H_ * H_;
  auto loadBn = [&](int ks) -> bf16x16 {
    int k = 8 * khalf + 32 * ks;
    BfVec v; const unsigned short* bp = wnT + (size_t)(n0 + lrow) * H_ + k;
    v.q[0] = *(const uint4*)(bp); v.q[1] = *(const uint4*)(bp + 16);
    return v.v;
  };

  // Preload loop-invariant u/r recurrent weights into VGPRs: 2 gates x 8 K-steps.
  bf16x16 wb[2][8];
  #pragma unroll
  for (int g = 0; g < 2; ++g) {
    const unsigned short* wT = whT + (size_t)g * H_ * H_;
    #pragma unroll
    for (int ks = 0; ks < 8; ++ks) {
      int k = 8 * khalf + 32 * ks;
      BfVec Bv;
      const unsigned short* bp = wT + (size_t)(n0 + lrow) * H_ + k;
      Bv.q[0] = *(const uint4*)(bp);
      Bv.q[1] = *(const uint4*)(bp + 16);
      wb[g][ks] = Bv.v;
    }
  }

  // Init h: LDS bf16 copy for matmuls + f32 register copy of this wave's own tile.
  for (int i = threadIdx.x; i < 16 * H_; i += 512) {
    int m = i >> 8, n = i & 255;
    hl[0][m * ROWP + n] = f2bf(h0[(size_t)(b0 + m) * H_ + n]);
  }
  float hreg[8];
  #pragma unroll
  for (int r = 0; r < 8; ++r)
    hreg[r] = h0[(size_t)(b0 + r + 8 * khalf) * H_ + n0 + lrow];
  __syncthreads();

  const float* xu = xg;
  const float* xr = xg + (size_t)T_ * B_ * H_;
  const float* xn = xg + 2 * (size_t)T_ * B_ * H_;

  for (int t = 0; t < T_; ++t) {
    const int cur = t & 1, nxt = cur ^ 1;

    // Issue staged-gate loads first: their HBM/L2 latency overlaps the WMMA chain.
    float xuv[8], xrv[8], xnv[8];
    size_t base = ((size_t)t * B_ + b0 + 8 * khalf) * H_ + n0 + lrow;
    #pragma unroll
    for (int r = 0; r < 8; ++r) {
      xuv[r] = xu[base + (size_t)r * H_];
      xrv[r] = xr[base + (size_t)r * H_];
      xnv[r] = xn[base + (size_t)r * H_];
    }
    if (t + 1 < T_) {                         // pull next step's rows toward L2/L0
      __builtin_prefetch(&xu[base + (size_t)B_ * H_], 0, 1);
      __builtin_prefetch(&xr[base + (size_t)B_ * H_], 0, 1);
      __builtin_prefetch(&xn[base + (size_t)B_ * H_], 0, 1);
    }

    bf16x16 bn4[4];
    bn4[0] = loadBn(0);
    bn4[1] = loadBn(1);
    bn4[2] = loadBn(2);

    f32x8 au = {}, ar = {}, an = {};
    #pragma unroll
    for (int ks = 0; ks < 8; ++ks) {
      if (ks < 5) bn4[(ks + 3) & 3] = loadBn(ks + 3);
      BfVec A;  // h tile from LDS
      const unsigned short* ap = &hl[cur][lrow * ROWP + 8 * khalf + 32 * ks];
      A.q[0] = *(const uint4*)(ap);
      A.q[1] = *(const uint4*)(ap + 16);
      au = __builtin_amdgcn_wmma_f32_16x16x32_bf16(false, A.v, false, wb[0][ks], (short)0, au, false, false);
      ar = __builtin_amdgcn_wmma_f32_16x16x32_bf16(false, A.v, false, wb[1][ks], (short)0, ar, false, false);
      an = __builtin_amdgcn_wmma_f32_16x16x32_bf16(false, A.v, false, bn4[ks & 3], (short)0, an, false, false);
    }

    #pragma unroll
    for (int r = 0; r < 8; ++r) {
      int M = r + 8 * khalf;
      int b = b0 + M;
      int n = n0 + lrow;
      float rt = 1.f / (1.f + __expf(-(xuv[r] + au[r])));
      float zt = 1.f / (1.f + __expf(-(xrv[r] + ar[r])));
      float nt = tanhf(xnv[r] + rt * an[r]);
      float hn = (1.f - zt) * nt + zt * hreg[r];
      hreg[r] = hn;
      out[((size_t)b * T_ + t) * H_ + n] = hn;            // hidden_states[b][t][n]
      hl[nxt][M * ROWP + n] = f2bf(hn);                   // next step's A operand
    }
    __syncthreads();
  }

  // ht = h_last
  #pragma unroll
  for (int r = 0; r < 8; ++r) {
    int b = b0 + r + 8 * khalf;
    out[(size_t)B_ * T_ * H_ + (size_t)b * H_ + n0 + lrow] = hreg[r];
  }
}

extern "C" void kernel_launch(void* const* d_in, const int* in_sizes, int n_in,
                              void* d_out, int out_size, void* d_ws, size_t ws_size,
                              hipStream_t stream) {
  (void)in_sizes; (void)n_in; (void)out_size; (void)ws_size;
  const float* x  = (const float*)d_in[0];
  const float* h0 = (const float*)d_in[1];
  const float* Wu = (const float*)d_in[2];
  const float* bu = (const float*)d_in[3];
  const float* Wr = (const float*)d_in[4];
  const float* br = (const float*)d_in[5];
  const float* Wn = (const float*)d_in[6];
  const float* bn = (const float*)d_in[7];
  float* out = (float*)d_out;

  char* w = (char*)d_ws;
  unsigned short* xb  = (unsigned short*)w;                    // B*T*E bf16
  size_t o1 = (size_t)B_ * T_ * E_ * 2;
  unsigned short* wxT = (unsigned short*)(w + o1);             // 3*E*H bf16 (transposed)
  size_t o2 = o1 + (size_t)3 * E_ * H_ * 2;
  unsigned short* whT = (unsigned short*)(w + o2);             // 3*H*H bf16 (transposed)
  size_t o3 = o2 + (size_t)3 * H_ * H_ * 2;
  float* xg = (float*)(w + o3);                                // 3*T*B*H f32 staged gates

  k_conv_x<<<dim3((B_ * T_ * E_) / (256 * 8)), dim3(256), 0, stream>>>(x, xb);
  k_conv_w<<<dim3((3 * H_ * (E_ + H_)) / 256), dim3(256), 0, stream>>>(Wu, Wr, Wn, wxT, whT);
  k_proj<<<dim3((B_ * T_) / 16 / 4, 3), dim3(128), 0, stream>>>(xb, wxT, bu, br, bn, xg);
  k_rnn<<<dim3(4), dim3(512), 0, stream>>>(whT, xg, h0, out);
}
